// Generator_authors_18098992185652
// MI455X (gfx1250) — matrix operand order repaired and checked by reading, not compiled, not run on hardware
//
#include <hip/hip_runtime.h>
#include <hip/hip_bf16.h>
#include <cstddef>
#include <cstdint>

// ============================================================================
// MI455X (gfx1250, wave32) implementation.
// Every conv / linear is an implicit GEMM on V_WMMA_F32_16X16X32_F16 (f16
// operands, f32 accumulation). Each wave owns a 16(M=Cout) x 64(N=pixels)
// macro-tile (4 accumulators). All waves in a block share the same M tile, so
// the 16x32 A (weight) fragment tile is staged into LDS once per block per
// K-chunk -- via global_load_async_to_lds_b128 + s_wait_asynccnt when the
// toolchain exposes them (__has_builtin-guarded), else plain loads + ds_store
// -- and each wave assembles its A fragment with two 16B ds_load_b128 reads.
// Kernel is templated on (KH,KW,STRIDE) so im2col uses constant divisors.
// BN stats, GRU gating and the latent head stay in f32.
// ============================================================================

typedef __attribute__((ext_vector_type(16))) _Float16 v16h;
typedef __attribute__((ext_vector_type(8)))  _Float16 v8h;
typedef __attribute__((ext_vector_type(8)))  float    v8f;

#define EPS_BN 1e-5f

#if __has_builtin(__builtin_amdgcn_global_load_async_to_lds_b128) && \
    __has_builtin(__builtin_amdgcn_s_wait_asynccnt)
#define HAVE_ASYNC_LDS 1
// Builtin signature (from hipcc diagnostic): param0 = v4i in AS(1) (global),
// param1 = v4i in AS(3) (LDS), then imm offset + imm cpol.
typedef int v4i_vs __attribute__((vector_size(16)));
typedef __attribute__((address_space(1))) v4i_vs* async_gptr;
typedef __attribute__((address_space(3))) v4i_vs* async_lptr;
#else
#define HAVE_ASYNC_LDS 0
#endif

static inline int ceil_div_i(int a, int b) { return (a + b - 1) / b; }

// LDS tile: 16 rows x 32 halves of A, row pitch 40 halves (20 DWORDs; odd
// stride mod 64 banks -> conflict-free 16B lane reads).
#define LDS_PITCH 40

// ---------------------------------------------------------------------------
// Implicit-GEMM conv: M = Cout, N = B*Hout*Wout, K = Cin*KH*KW.
// Linear layers use the <1,1,1> instantiation (H=W=Hout=Wout=1).
// ---------------------------------------------------------------------------
template <int KH, int KW, int STRIDE>
__global__ void conv_wmma_k(const _Float16* __restrict__ X,
                            const _Float16* __restrict__ Wt,   // [Cout, Cin*KH*KW]
                            float* __restrict__ Y,             // [B, Cout, Hout, Wout] f32
                            int Bt, int Cin, int H, int Wd,
                            int Cout, int Hout, int Wout, int pad)
{
  __shared__ _Float16 ldsA[16 * LDS_PITCH];

  const int tid   = threadIdx.x;
  const int lane  = tid & 31;
  const int wave  = tid >> 5;
  const int waves = blockDim.x >> 5;
  const int N     = Bt * Hout * Wout;
  const int K     = Cin * KH * KW;
  const int sTile = blockIdx.x * waves + wave;   // 64-column super tile
  const int n0    = sTile * 64;
  const int m0    = blockIdx.y * 16;             // shared by all waves in block

  const int laneLo = lane & 15;
  const int laneHi = lane >> 4;

  // NOTE: no early exit -- barriers inside the K loop require all waves.
  int wo[4], ho[4], bi[4]; bool ok[4];
#pragma unroll
  for (int s = 0; s < 4; ++s) {
    int n = n0 + s * 16 + laneLo;
    ok[s] = (n < N);
    int ns = ok[s] ? n : 0;
    wo[s] = ns % Wout;
    int t = ns / Wout;
    ho[s] = t % Hout;
    bi[s] = t / Hout;
  }

  const bool kVec = ((K & 7) == 0);              // 16B-aligned weight rows

  v8f acc[4];
#pragma unroll
  for (int s = 0; s < 4; ++s) acc[s] = (v8f){0.f,0.f,0.f,0.f,0.f,0.f,0.f,0.f};

  for (int kc = 0; kc < K; kc += 32) {
    const bool full = (kc + 32 <= K);

    // ---- stage A tile (16 rows x 32 halves) into LDS, once per block ----
#if HAVE_ASYNC_LDS
    if (full && kVec && (m0 + 16) <= Cout) {
      for (int ch = tid; ch < 64; ch += blockDim.x) {       // 64 x 16B chunks
        int row = ch >> 2, c = (ch & 3) * 8;
        const _Float16* src = Wt + (size_t)(m0 + row) * K + kc + c;
        __builtin_amdgcn_global_load_async_to_lds_b128(
            (async_gptr)src,
            (async_lptr)&ldsA[row * LDS_PITCH + c],
            0, 0);
      }
      __builtin_amdgcn_s_wait_asynccnt(0);
    } else
#endif
    {
      for (int idx = tid * 4; idx < 512; idx += blockDim.x * 4) {
        int row = idx >> 5, col = idx & 31;
        const _Float16* src = Wt + (size_t)(m0 + row) * K + kc + col;
        bool rOk = (m0 + row) < Cout;
#pragma unroll
        for (int u = 0; u < 4; ++u) {
          int k = kc + col + u;
          ldsA[row * LDS_PITCH + col + u] = (rOk && k < K) ? src[u] : (_Float16)0.f;
        }
      }
    }
    __syncthreads();

    // ---- A fragment from LDS: two contiguous 8-half runs per lane ----
    v16h afrag;
    {
      v8h alo = *(const v8h*)&ldsA[laneLo * LDS_PITCH + laneHi * 16];
      v8h ahi = *(const v8h*)&ldsA[laneLo * LDS_PITCH + laneHi * 16 + 8];
#pragma unroll
      for (int h = 0; h < 8; ++h) { afrag[h] = alo[h]; afrag[h + 8] = ahi[h]; }
    }

    // ---- B fragments: lane = column, k = kc + laneHi*16 + h ----
    v16h bfrag[4];
    if constexpr (KH == 1 && KW == 1) {
#pragma unroll
      for (int s = 0; s < 4; ++s) {
        int n = n0 + s * 16 + laneLo;
        if (ok[s] && full && kVec) {
          const v8h* pb = (const v8h*)(X + (size_t)n * K + kc + laneHi * 16);
          v8h b0 = pb[0];
          v8h b1 = pb[1];
#pragma unroll
          for (int h = 0; h < 8; ++h) { bfrag[s][h] = b0[h]; bfrag[s][h + 8] = b1[h]; }
        } else {
#pragma unroll
          for (int h = 0; h < 16; ++h) {
            int k = kc + laneHi * 16 + h;
            bfrag[s][h] = (ok[s] && k < K) ? X[(size_t)n * K + k] : (_Float16)0.f;
          }
        }
      }
    } else {
#pragma unroll
      for (int h = 0; h < 16; ++h) {
        int k = kc + laneHi * 16 + h;
        if (k < K) {
          int kw = k % KW;            // constant divisors -> cheap
          int t2 = k / KW;
          int kh = t2 % KH;
          int ci = t2 / KH;
#pragma unroll
          for (int s = 0; s < 4; ++s) {
            int hi = ho[s] * STRIDE - pad + kh;
            int wi = wo[s] * STRIDE - pad + kw;
            _Float16 v = (_Float16)0.f;
            if (ok[s] && hi >= 0 && hi < H && wi >= 0 && wi < Wd)
              v = X[(((size_t)bi[s] * Cin + ci) * H + hi) * Wd + wi];
            bfrag[s][h] = v;
          }
        } else {
#pragma unroll
          for (int s = 0; s < 4; ++s) bfrag[s][h] = (_Float16)0.f;
        }
      }
    }

    acc[0] = __builtin_amdgcn_wmma_f32_16x16x32_f16(false, afrag, false, bfrag[0], (short)0, acc[0], false, false);
    acc[1] = __builtin_amdgcn_wmma_f32_16x16x32_f16(false, afrag, false, bfrag[1], (short)0, acc[1], false, false);
    acc[2] = __builtin_amdgcn_wmma_f32_16x16x32_f16(false, afrag, false, bfrag[2], (short)0, acc[2], false, false);
    acc[3] = __builtin_amdgcn_wmma_f32_16x16x32_f16(false, afrag, false, bfrag[3], (short)0, acc[3], false, false);

    __syncthreads();   // before next chunk overwrites ldsA
  }

  // C/D layout: VGPR r, lanes 0-15 -> M=r, N=lane; lanes 16-31 -> M=8+r.
#pragma unroll
  for (int s = 0; s < 4; ++s) {
    if (!ok[s]) continue;
#pragma unroll
    for (int r = 0; r < 8; ++r) {
      int m = m0 + r + laneHi * 8;
      if (m < Cout)
        Y[(((size_t)bi[s] * Cout + m) * Hout + ho[s]) * Wout + wo[s]] = acc[s][r];
    }
  }
}

// ---------------------------------------------------------------------------
// Elementwise / reduction kernels
// ---------------------------------------------------------------------------
__device__ __forceinline__ float sigmoid_(float x) { return 1.0f / (1.0f + expf(-x)); }

__global__ void cast_f32_to_f16_k(const float* __restrict__ x, _Float16* __restrict__ y, int n) {
  int i = blockIdx.x * blockDim.x + threadIdx.x;
  if (i < n) y[i] = (_Float16)x[i];
}

__global__ void zero_h_k(float* h, _Float16* hh, int n) {
  int i = blockIdx.x * blockDim.x + threadIdx.x;
  if (i < n) { h[i] = 0.f; hh[i] = (_Float16)0.f; }
}

__global__ void add_f16_k(const _Float16* a, const _Float16* b, _Float16* y, int n) {
  int i = blockIdx.x * blockDim.x + threadIdx.x;
  if (i < n) y[i] = (_Float16)((float)a[i] + (float)b[i]);
}

__global__ void up2_k(const _Float16* __restrict__ x, _Float16* __restrict__ y,
                      int B, int C, int H, int W) {
  int Ho = H * 2, Wo = W * 2;
  int total = B * C * Ho * Wo;
  int i = blockIdx.x * blockDim.x + threadIdx.x;
  if (i >= total) return;
  int wo = i % Wo; int t = i / Wo;
  int ho = t % Ho; t /= Ho;
  int c  = t % C;  int b = t / C;
  y[i] = x[((size_t)(b * C + c) * H + (ho >> 1)) * W + (wo >> 1)];
}

__global__ void concat_bcast_k(const _Float16* __restrict__ e, const _Float16* __restrict__ c,
                               _Float16* __restrict__ out, int B, int Ce, int Cc, int HW) {
  int total = B * (Ce + Cc) * HW;
  int i = blockIdx.x * blockDim.x + threadIdx.x;
  if (i >= total) return;
  int s = i % HW; int t = i / HW;
  int ch = t % (Ce + Cc); int b = t / (Ce + Cc);
  out[i] = (ch < Ce) ? e[((size_t)(b * Ce + ch)) * HW + s] : c[b * Cc + (ch - Ce)];
}

__global__ void bias_relu_k(const float* __restrict__ x, const float* __restrict__ bias,
                            _Float16* __restrict__ y, int C, int HW, int total) {
  int i = blockIdx.x * blockDim.x + threadIdx.x;
  if (i >= total) return;
  int c = (i / HW) % C;
  y[i] = (_Float16)fmaxf(x[i] + bias[c], 0.f);
}

__global__ void bias_tanh_k(const float* __restrict__ x, const float* __restrict__ bias,
                            float* __restrict__ y, int C, int HW, int total) {
  int i = blockIdx.x * blockDim.x + threadIdx.x;
  if (i >= total) return;
  int c = (i / HW) % C;
  y[i] = tanhf(x[i] + bias[c]);
}

__global__ void bn_stats_k(const float* __restrict__ x, float* __restrict__ mean,
                           float* __restrict__ var, int B, int C, int HW) {
  int c = blockIdx.x, tid = threadIdx.x;
  int n = B * HW;
  float s = 0.f, s2 = 0.f;
  for (int i = tid; i < n; i += blockDim.x) {
    int b = i / HW, sp = i - b * HW;
    float v = x[((size_t)b * C + c) * HW + sp];
    s += v; s2 += v * v;
  }
  __shared__ float sh[256], sh2[256];
  sh[tid] = s; sh2[tid] = s2; __syncthreads();
  for (int o = 128; o > 0; o >>= 1) {
    if (tid < o) { sh[tid] += sh[tid + o]; sh2[tid] += sh2[tid + o]; }
    __syncthreads();
  }
  if (tid == 0) {
    float m = sh[0] / n;
    mean[c] = m;
    var[c]  = sh2[0] / n - m * m;   // biased, matches jnp.var
  }
}

__global__ void bn_apply_k(const float* __restrict__ x, const float* __restrict__ mean,
                           const float* __restrict__ var, const float* __restrict__ g,
                           const float* __restrict__ bb, const _Float16* __restrict__ res,
                           _Float16* __restrict__ y, int C, int HW, int total, int relu) {
  int i = blockIdx.x * blockDim.x + threadIdx.x;
  if (i >= total) return;
  int c = (i / HW) % C;
  float v = (x[i] - mean[c]) * rsqrtf(var[c] + EPS_BN) * g[c] + bb[c];
  if (res)  v += (float)res[i];
  if (relu) v = fmaxf(v, 0.f);
  y[i] = (_Float16)v;
}

// One GRU step (r/z/n gating) + optional backward mask blend.
__global__ void gru_step_k(const float* __restrict__ gx, const float* __restrict__ gh,
                           const float* __restrict__ bih, const float* __restrict__ bhh,
                           float* __restrict__ hcur, _Float16* __restrict__ hcurh,
                           float* __restrict__ hsave, const int* __restrict__ len,
                           int t, int T, int useMask, int B, int Hd) {
  int i = blockIdx.x * blockDim.x + threadIdx.x;
  if (i >= B * Hd) return;
  int b = i / Hd, j = i - b * Hd;
  int g0 = b * 3 * Hd + j;
  float r  = sigmoid_(gx[g0]          + bih[j]          + gh[g0]          + bhh[j]);
  float z  = sigmoid_(gx[g0 + Hd]     + bih[Hd + j]     + gh[g0 + Hd]     + bhh[Hd + j]);
  float nv = tanhf  (gx[g0 + 2 * Hd]  + bih[2 * Hd + j] + r * (gh[g0 + 2 * Hd] + bhh[2 * Hd + j]));
  float hp = hcur[i];
  float h  = (1.f - z) * nv + z * hp;
  if (useMask) {
    float m = ((T - 1 - t) < len[b]) ? 1.f : 0.f;
    h = m * h + (1.f - m) * hp;
  }
  hcur[i] = h; hcurh[i] = (_Float16)h; hsave[i] = h;
}

// h = (hF * mask[::-1] + hB[::-1]) * 0.5 ; cond = sum_t h / len[b]
__global__ void combine_cond_k(const float* __restrict__ hF, const float* __restrict__ hB,
                               const int* __restrict__ len, float* __restrict__ cond,
                               _Float16* __restrict__ condh, int T, int B, int Hd) {
  int i = blockIdx.x * blockDim.x + threadIdx.x;
  if (i >= B * Hd) return;
  int b = i / Hd;
  float s = 0.f;
  for (int t = 0; t < T; ++t) {
    float mrev = (t < len[b]) ? 1.f : 0.f;          // mask[T-1-t][b]
    s += (hF[(size_t)t * B * Hd + i] * mrev + hB[(size_t)(T - 1 - t) * B * Hd + i]) * 0.5f;
  }
  float c = s / (float)len[b];
  cond[i] = c; condh[i] = (_Float16)c;
}

// z_mean/z_log = leaky_relu(raw + bias, 0.2); c = z_mean + exp(z_log)*z
__global__ void latent_k(const float* __restrict__ zmraw, const float* __restrict__ zlraw,
                         const float* __restrict__ mub, const float* __restrict__ lsb,
                         const float* __restrict__ z, float* __restrict__ outzm,
                         float* __restrict__ outzl, _Float16* __restrict__ ch,
                         int B, int L) {
  int i = blockIdx.x * blockDim.x + threadIdx.x;
  if (i >= B * L) return;
  int j = i % L;
  float zm = zmraw[i] + mub[j]; zm = (zm > 0.f) ? zm : 0.2f * zm;
  float zl = zlraw[i] + lsb[j]; zl = (zl > 0.f) ? zl : 0.2f * zl;
  outzm[i] = zm; outzl[i] = zl;
  ch[i] = (_Float16)(zm + expf(zl) * z[i]);
}

// ---------------------------------------------------------------------------
// Host orchestration
// ---------------------------------------------------------------------------
enum {
  IN_IMG = 0, IN_TXT, IN_LEN, IN_Z,
  P_E0_W, P_E0_B, P_E1_W, P_E1_G, P_E1_B, P_E2_W, P_E2_G, P_E2_B,
  P_E3_W, P_E3_G, P_E3_B, P_R_W, P_R_G, P_R_B,
  P_D1_W, P_D1_G, P_D1_B, P_D2_W, P_D2_G, P_D2_B, P_D3_W, P_D3_G, P_D3_B,
  P_D4_W, P_D4_B, P_MU_W, P_MU_B, P_LS_W, P_LS_B,
  P_F_WIH, P_F_WHH, P_F_BIH, P_F_BHH, P_B_WIH, P_B_WHH, P_B_BIH, P_B_BHH,
  P_RB0   // rb{i}: {w1,g1,b1,w2,g2,b2} at P_RB0 + 6*i
};

extern "C" void kernel_launch(void* const* d_in, const int* in_sizes, int n_in,
                              void* d_out, int out_size, void* d_ws, size_t ws_size,
                              hipStream_t stream)
{
  (void)in_sizes; (void)n_in; (void)out_size; (void)ws_size;
  const int B = 16, T = 24, TXTD = 300, HD = 512, G3 = 1536, LAT = 128;

  const float* img = (const float*)d_in[IN_IMG];
  const float* txt = (const float*)d_in[IN_TXT];
  const int*   len = (const int*)  d_in[IN_LEN];
  const float* zin = (const float*)d_in[IN_Z];
  float* out = (float*)d_out;                  // [16*3*128*128 | 16*128 | 16*128]

  auto P = [&](int i) -> const float* { return (const float*)d_in[i]; };

  // ---- workspace bump allocator ----
  char* wsb = (char*)d_ws; size_t off = 0;
  auto allocB = [&](size_t bytes) -> void* {
    off = (off + 255) & ~(size_t)255;
    void* p = wsb + off; off += bytes; return p;
  };
  auto aH = [&](size_t n) -> _Float16* { return (_Float16*)allocB(n * sizeof(_Float16)); };
  auto aF = [&](size_t n) -> float*    { return (float*)   allocB(n * sizeof(float)); };

  auto cast16 = [&](const float* s, _Float16* d, int n) {
    cast_f32_to_f16_k<<<ceil_div_i(n, 256), 256, 0, stream>>>(s, d, n);
  };
  auto conv = [&](const _Float16* X, const _Float16* Wt, float* Y,
                  int Bt, int Cin, int H, int Wd, int Cout, int Hout, int Wout,
                  int KH, int KW, int st, int pad) {
    int N  = Bt * Hout * Wout;
    int sT = ceil_div_i(N, 64), mT = ceil_div_i(Cout, 16);
    int waves = sT < 4 ? sT : 4;
    dim3 grid(ceil_div_i(sT, waves), mT);
    dim3 blk(waves * 32);
    if (KH == 3 && KW == 3 && st == 1)
      conv_wmma_k<3,3,1><<<grid, blk, 0, stream>>>(X, Wt, Y, Bt, Cin, H, Wd, Cout, Hout, Wout, pad);
    else if (KH == 4 && KW == 4 && st == 2)
      conv_wmma_k<4,4,2><<<grid, blk, 0, stream>>>(X, Wt, Y, Bt, Cin, H, Wd, Cout, Hout, Wout, pad);
    else
      conv_wmma_k<1,1,1><<<grid, blk, 0, stream>>>(X, Wt, Y, Bt, Cin, H, Wd, Cout, Hout, Wout, pad);
  };
  auto gemm = [&](const _Float16* X, const _Float16* Wt, float* Y, int rows, int K, int M) {
    conv(X, Wt, Y, rows, K, 1, 1, M, 1, 1, 1, 1, 1, 0);
  };

  float* meanb = aF(512);
  float* varb  = aF(512);
  auto bn = [&](const float* xin, const float* g, const float* bb, const _Float16* res,
                _Float16* yout, int C, int HW, int relu) {
    bn_stats_k<<<C, 256, 0, stream>>>(xin, meanb, varb, B, C, HW);
    int total = B * C * HW;
    bn_apply_k<<<ceil_div_i(total, 256), 256, 0, stream>>>(xin, meanb, varb, g, bb, res,
                                                           yout, C, HW, total, relu);
  };

  // ---- f16 weight copies ----
  _Float16* wE0 = aH(64 * 27);          cast16(P(P_E0_W), wE0, 64 * 27);
  _Float16* wE1 = aH(128 * 1024);       cast16(P(P_E1_W), wE1, 128 * 1024);
  _Float16* wE2 = aH(256 * 2048);       cast16(P(P_E2_W), wE2, 256 * 2048);
  _Float16* wE3 = aH(512 * 4096);       cast16(P(P_E3_W), wE3, 512 * 4096);
  _Float16* wR  = aH(512 * 5760);       cast16(P(P_R_W),  wR,  512 * 5760);
  _Float16* wRB[8];
  for (int i = 0; i < 4; ++i) {
    wRB[2*i]   = aH(512 * 4608); cast16(P(P_RB0 + 6*i + 0), wRB[2*i],   512 * 4608);
    wRB[2*i+1] = aH(512 * 4608); cast16(P(P_RB0 + 6*i + 3), wRB[2*i+1], 512 * 4608);
  }
  _Float16* wD1 = aH(256 * 4608);       cast16(P(P_D1_W), wD1, 256 * 4608);
  _Float16* wD2 = aH(128 * 2304);       cast16(P(P_D2_W), wD2, 128 * 2304);
  _Float16* wD3 = aH(64 * 1152);        cast16(P(P_D3_W), wD3, 64 * 1152);
  _Float16* wD4 = aH(3 * 576);          cast16(P(P_D4_W), wD4, 3 * 576);
  _Float16* wMU = aH(128 * 512);        cast16(P(P_MU_W), wMU, 128 * 512);
  _Float16* wLS = aH(128 * 512);        cast16(P(P_LS_W), wLS, 128 * 512);
  _Float16* wFI = aH(G3 * TXTD);        cast16(P(P_F_WIH), wFI, G3 * TXTD);
  _Float16* wFH = aH(G3 * HD);          cast16(P(P_F_WHH), wFH, G3 * HD);
  _Float16* wBI = aH(G3 * TXTD);        cast16(P(P_B_WIH), wBI, G3 * TXTD);
  _Float16* wBH = aH(G3 * HD);          cast16(P(P_B_WHH), wBH, G3 * HD);

  // ---- activation / scratch buffers ----
  _Float16* imgH  = aH(16 * 3 * 128 * 128);
  _Float16* e0act = aH(16 * 64 * 128 * 128);   // reused later: upb2, then d3act
  _Float16* e1act = aH(16 * 128 * 64 * 64);    // reused later: up1, then d2act
  _Float16* e2act = aH(16 * 256 * 32 * 32);    // reused later: d1act
  _Float16* e3act = aH(16 * 512 * 16 * 16);
  _Float16* catH  = aH(16 * 640 * 256);
  _Float16* m0A   = aH(16 * 512 * 256);
  _Float16* m0B   = aH(16 * 512 * 256);
  _Float16* t1act = aH(16 * 512 * 256);
  _Float16* dsum  = aH(16 * 512 * 256);
  _Float16* up3   = aH(16 * 128 * 128 * 128);
  float*    convout = aF(16 * 64 * 128 * 128); // biggest f32 conv output, reused per conv

  _Float16* txtH  = aH(T * B * TXTD);
  float* gxF   = aF((size_t)T * B * G3);
  float* gxB   = aF((size_t)T * B * G3);
  float* ghbuf = aF(B * G3);
  float* hFst  = aF((size_t)T * B * HD);
  float* hBst  = aF((size_t)T * B * HD);
  float* hcur  = aF(B * HD);
  _Float16* hcurH = aH(B * HD);
  float* cond  = aF(B * HD);
  _Float16* condH = aH(B * HD);
  float* zmraw = aF(B * LAT);
  float* zlraw = aF(B * LAT);
  _Float16* cH = aH(B * LAT);

  // ========================= encoder =========================
  cast16(img, imgH, 16 * 3 * 128 * 128);
  conv(imgH, wE0, convout, B, 3, 128, 128, 64, 128, 128, 3, 3, 1, 1);
  { int tot = 16 * 64 * 128 * 128;
    bias_relu_k<<<ceil_div_i(tot, 256), 256, 0, stream>>>(convout, P(P_E0_B), e0act, 64, 128 * 128, tot); }
  conv(e0act, wE1, convout, B, 64, 128, 128, 128, 64, 64, 4, 4, 2, 1);
  bn(convout, P(P_E1_G), P(P_E1_B), nullptr, e1act, 128, 64 * 64, 1);
  conv(e1act, wE2, convout, B, 128, 64, 64, 256, 32, 32, 4, 4, 2, 1);
  bn(convout, P(P_E2_G), P(P_E2_B), nullptr, e2act, 256, 32 * 32, 1);
  conv(e2act, wE3, convout, B, 256, 32, 32, 512, 16, 16, 4, 4, 2, 1);
  bn(convout, P(P_E3_G), P(P_E3_B), nullptr, e3act, 512, 16 * 16, 1);

  // ========================= bidirectional GRU =========================
  cast16(txt, txtH, T * B * TXTD);
  gemm(txtH, wFI, gxF, T * B, TXTD, G3);     // all forward input projections
  gemm(txtH, wBI, gxB, T * B, TXTD, G3);     // all backward input projections

  zero_h_k<<<ceil_div_i(B * HD, 256), 256, 0, stream>>>(hcur, hcurH, B * HD);
  for (int t = 0; t < T; ++t) {
    gemm(hcurH, wFH, ghbuf, B, HD, G3);
    gru_step_k<<<ceil_div_i(B * HD, 256), 256, 0, stream>>>(
        gxF + (size_t)t * B * G3, ghbuf, P(P_F_BIH), P(P_F_BHH),
        hcur, hcurH, hFst + (size_t)t * B * HD, len, t, T, 0, B, HD);
  }
  zero_h_k<<<ceil_div_i(B * HD, 256), 256, 0, stream>>>(hcur, hcurH, B * HD);
  for (int t = 0; t < T; ++t) {
    gemm(hcurH, wBH, ghbuf, B, HD, G3);
    gru_step_k<<<ceil_div_i(B * HD, 256), 256, 0, stream>>>(
        gxB + (size_t)(T - 1 - t) * B * G3, ghbuf, P(P_B_BIH), P(P_B_BHH),
        hcur, hcurH, hBst + (size_t)t * B * HD, len, t, T, 1, B, HD);
  }
  combine_cond_k<<<ceil_div_i(B * HD, 256), 256, 0, stream>>>(hFst, hBst, len, cond, condH, T, B, HD);

  // ========================= latent head =========================
  gemm(condH, wMU, zmraw, B, HD, LAT);
  gemm(condH, wLS, zlraw, B, HD, LAT);
  latent_k<<<ceil_div_i(B * LAT, 256), 256, 0, stream>>>(
      zmraw, zlraw, P(P_MU_B), P(P_LS_B), zin,
      out + 16 * 3 * 128 * 128, out + 16 * 3 * 128 * 128 + B * LAT, cH, B, LAT);

  // ========================= fusion + resblocks =========================
  { int tot = 16 * 640 * 256;
    concat_bcast_k<<<ceil_div_i(tot, 256), 256, 0, stream>>>(e3act, cH, catH, B, 512, 128, 256); }
  conv(catH, wR, convout, B, 640, 16, 16, 512, 16, 16, 3, 3, 1, 1);
  bn(convout, P(P_R_G), P(P_R_B), nullptr, m0A, 512, 256, 1);

  _Float16* xcur = m0A;
  for (int i = 0; i < 4; ++i) {
    conv(xcur, wRB[2*i], convout, B, 512, 16, 16, 512, 16, 16, 3, 3, 1, 1);
    bn(convout, P(P_RB0 + 6*i + 1), P(P_RB0 + 6*i + 2), nullptr, t1act, 512, 256, 1);
    conv(t1act, wRB[2*i+1], convout, B, 512, 16, 16, 512, 16, 16, 3, 3, 1, 1);
    _Float16* xnext = (xcur == m0A) ? m0B : m0A;
    bn(convout, P(P_RB0 + 6*i + 4), P(P_RB0 + 6*i + 5), xcur, xnext, 512, 256, 0);
    xcur = xnext;
  }
  add_f16_k<<<ceil_div_i(16 * 512 * 256, 256), 256, 0, stream>>>(e3act, xcur, dsum, 16 * 512 * 256);

  // ========================= decoder (buffers aliased onto dead encoder acts) =====
  _Float16* up1   = e1act;   // 16*512*32*32  == 16*128*64*64 elems
  _Float16* d1act = e2act;   // 16*256*32*32
  _Float16* upb2  = e0act;   // 16*256*64*64  == 16*64*128*128
  _Float16* d2act = e1act;   // (up1 dead by then)
  _Float16* d3act = e0act;   // (upb2 dead by then)

  up2_k<<<ceil_div_i(16 * 512 * 32 * 32, 256), 256, 0, stream>>>(dsum, up1, B, 512, 16, 16);
  conv(up1, wD1, convout, B, 512, 32, 32, 256, 32, 32, 3, 3, 1, 1);
  bn(convout, P(P_D1_G), P(P_D1_B), nullptr, d1act, 256, 32 * 32, 1);

  up2_k<<<ceil_div_i(16 * 256 * 64 * 64, 256), 256, 0, stream>>>(d1act, upb2, B, 256, 32, 32);
  conv(upb2, wD2, convout, B, 256, 64, 64, 128, 64, 64, 3, 3, 1, 1);
  bn(convout, P(P_D2_G), P(P_D2_B), nullptr, d2act, 128, 64 * 64, 1);

  up2_k<<<ceil_div_i(16 * 128 * 128 * 128, 256), 256, 0, stream>>>(d2act, up3, B, 128, 64, 64);
  conv(up3, wD3, convout, B, 128, 128, 128, 64, 128, 128, 3, 3, 1, 1);
  bn(convout, P(P_D3_G), P(P_D3_B), nullptr, d3act, 64, 128 * 128, 1);

  conv(d3act, wD4, convout, B, 64, 128, 128, 3, 128, 128, 3, 3, 1, 1);
  { int tot = 16 * 3 * 128 * 128;
    bias_tanh_k<<<ceil_div_i(tot, 256), 256, 0, stream>>>(convout, P(P_D4_B), out, 3, 128 * 128, tot); }
}
